// RGBuvHistBlock_35278861369608
// MI455X (gfx1250) — compile-verified, matrix operand order checked by reading, not run on hardware
//
#include <hip/hip_runtime.h>
#include <hip/hip_bf16.h>

typedef __attribute__((ext_vector_type(16))) _Float16 v16h;
typedef __attribute__((ext_vector_type(8)))  float    v8f;

#define H_BINS        64
#define HW            65536      // 256*256
#define NCH           24         // 8 * 3
#define BLOCKS_PER_CH 16
#define THREADS       256        // 8 waves (wave32)
#define WAVES         8
#define PIX_PER_BLOCK (HW / BLOCKS_PER_CH)    // 4096
#define PIX_PER_WAVE  (PIX_PER_BLOCK / WAVES) // 512
#define TILES         (PIX_PER_WAVE / 32)     // 16 tiles of 32 pixels

// exp(-d^2/(2*sigma^2)) with d = 2x-1-v_b, sigma = 0.02:
//   = exp2(-(e*e)),  e = x*(2*S) - S*(1+v_b),  S = sqrt(log2(e)/(2*sigma^2))
#define GAUSS_S   42.466089f   // sqrt(1803.3688)
#define GAUSS_2S  84.932178f

// ---------------------------------------------------------------------------
// Kernel 1: zero the output accumulator (d_out is poisoned by the harness).
// ---------------------------------------------------------------------------
__global__ void zero_out_kernel(float* __restrict__ out, int n) {
    int i = blockIdx.x * blockDim.x + threadIdx.x;
    if (i < n) out[i] = 0.0f;
}

// ---------------------------------------------------------------------------
// Kernel 2: WMMA soft-histogram accumulation.
//
// Per wave: iterate 32-pixel tiles. For each tile build B (32 pixels x 16
// bins, f16) per bin-group and accumulate D = ones(16x32) * B + D with
// v_wmma_f32_16x16x32_f16. Every row of D is the column (per-bin) sum of B;
// row M=0 lives in accumulator element 0 on lanes 0..15 (N = lane).
//
// Per element per group: v_pk_fma_f32 + v_pk_mul_f32(neg) + v_exp_f32 +
// packed f16 cvt -- the affine map and the Gaussian scale are fully folded.
// ---------------------------------------------------------------------------
__global__ __launch_bounds__(THREADS)
void hist_wmma_kernel(const float* __restrict__ x,
                      const float* __restrict__ binv,
                      float* __restrict__ out) {
    const int ch     = blockIdx.y;                 // channel: 0..23
    const int lane   = threadIdx.x & 31;
    const int wave   = threadIdx.x >> 5;
    const int half16 = lane & 16;                  // 0 for lanes 0-15, 16 for 16-31

    // Per-lane hoisted constant: cg[g] = -S * (1 + v_bin), bin N = lane&15.
    float cg[4];
#pragma unroll
    for (int g = 0; g < 4; ++g)
        cg[g] = -GAUSS_S * (1.0f + binv[g * 16 + (lane & 15)]);

    // A = all-ones 16x32 f16 matrix.
    v16h a;
#pragma unroll
    for (int i = 0; i < 16; ++i) a[i] = (_Float16)1.0f;

    v8f c[4];
#pragma unroll
    for (int g = 0; g < 4; ++g) c[g] = (v8f){};

    const float* base = x + (size_t)ch * HW
                          + (size_t)blockIdx.x * PIX_PER_BLOCK
                          + (size_t)wave * PIX_PER_WAVE;

    for (int t = 0; t < TILES; ++t) {
        const float* p = base + t * 32 + half16;   // 16 contiguous pixels
        if (t + 2 < TILES)                         // gfx1250 global_prefetch_b8
            __builtin_prefetch(base + (t + 2) * 32 + half16, 0, 1);

        // Raw pixel values; affine map is folded into the per-group FMA.
        float xs[16];
#pragma unroll
        for (int i = 0; i < 4; ++i) {
            float4 v = ((const float4*)p)[i];
            xs[4 * i + 0] = v.x;
            xs[4 * i + 1] = v.y;
            xs[4 * i + 2] = v.z;
            xs[4 * i + 3] = v.w;
        }

        // For each 16-bin group: build B (16 pixel-rows in this lane's bin
        // column) and fold 32 pixels into the f32 accumulator via WMMA.
#pragma unroll
        for (int g = 0; g < 4; ++g) {
            v16h b;
#pragma unroll
            for (int k = 0; k < 16; ++k) {
                float e = __builtin_fmaf(xs[k], GAUSS_2S, cg[g]);
                b[k] = (_Float16)__builtin_amdgcn_exp2f(-e * e);
            }
            c[g] = __builtin_amdgcn_wmma_f32_16x16x32_f16(
                /*neg_a=*/false, a, /*neg_b=*/false, b,
                /*c_mod=*/(short)0, c[g], /*reuse_a=*/false, /*reuse_b=*/false);
        }
    }

    // Row M=0 of each accumulator: element 0 on lanes 0..15 (N = lane).
    if (lane < 16) {
#pragma unroll
        for (int g = 0; g < 4; ++g)
            atomicAdd(&out[ch * H_BINS + g * 16 + lane], c[g][0]);
    }
}

// ---------------------------------------------------------------------------
// Kernel 3: per-channel L1 normalization. One block (64 threads) per channel.
// ---------------------------------------------------------------------------
__global__ void normalize_kernel(float* __restrict__ out) {
    __shared__ float s[H_BINS];
    const int ch = blockIdx.x;
    const int i  = threadIdx.x;

    float v = out[ch * H_BINS + i];
    s[i] = v;
    __syncthreads();
#pragma unroll
    for (int off = 32; off > 0; off >>= 1) {
        if (i < off) s[i] += s[i + off];
        __syncthreads();
    }
    out[ch * H_BINS + i] = v / (s[0] + 1e-8f);
}

// ---------------------------------------------------------------------------
extern "C" void kernel_launch(void* const* d_in, const int* in_sizes, int n_in,
                              void* d_out, int out_size, void* d_ws, size_t ws_size,
                              hipStream_t stream) {
    const float* x    = (const float*)d_in[0];   // [8,3,256,256] f32
    const float* binv = (const float*)d_in[1];   // [64] f32
    float*       out  = (float*)d_out;           // [24,64] f32

    zero_out_kernel<<<(NCH * H_BINS + 255) / 256, 256, 0, stream>>>(out, NCH * H_BINS);

    dim3 grid(BLOCKS_PER_CH, NCH);
    hist_wmma_kernel<<<grid, THREADS, 0, stream>>>(x, binv, out);

    normalize_kernel<<<NCH, H_BINS, 0, stream>>>(out);
}
